// GraphSAGEClassifier_3324304687696
// MI455X (gfx1250) — matrix operand order repaired and checked by reading, not compiled, not run on hardware
//
#include <hip/hip_runtime.h>
#include <hip/hip_bf16.h>

// ---------------------------------------------------------------------------
// GraphSAGE (project=True, mean aggr, L2-normalize) + linear classifier
// N=100000 nodes, E=1600000 edges, D=128 channels, 6 classes.
//
// Pipeline (all on `stream`):
//   0) hipMemsetAsync : zero summed (N*128 f32) + cnt (N f32)
//   1) k_pack : pre-pack Wp/Wl/Wr (f32 row-major) -> bf16 WMMA-fragment order
//   2) k_xp   : xp = relu(x @ Wp + bp), stored bf16            [WMMA bf16]
//   3) k_scat : scatter-mean numerator/denominator, f32 atomics (L2-resident)
//   4) k_out  : out = norm(agg@Wl + bl + xp@Wr) -> relu -> @Wf + bf
//               (two fused WMMAs per K-chunk, LDS row-norm reduction)
// ---------------------------------------------------------------------------

typedef __attribute__((ext_vector_type(16))) __bf16 v16bf;
typedef __attribute__((ext_vector_type(8)))  __bf16 v8bf;
typedef __attribute__((ext_vector_type(8)))  float  v8f;

#define NODES 100000
#define DCH   128
#define NCLS  6

__device__ __forceinline__ float bf16bits_to_f32(unsigned int hi16) {
    return __builtin_bit_cast(float, hi16 << 16);
}

// -------------------------------- pack -------------------------------------
// Pack a 128x128 f32 row-major weight into bf16 fragments:
//   frag[(tile*4 + chunk)*32 + lane][e] , e=2j+p -> W[K][col]
//   K = chunk*32 + 8h + 2j + p + 8*(j>=4) , col = tile*16 + (lane&15), h = lane>>4.
// 3 weights x 1024 fragments x 32B. One thread per fragment.
__global__ void k_pack(const float* __restrict__ Wp, const float* __restrict__ Wl,
                       const float* __restrict__ Wr, v16bf* __restrict__ pack) {
    const int id = blockIdx.x * blockDim.x + threadIdx.x;   // 0..3071
    const int w  = id >> 10;
    const int f  = id & 1023;
    const float* W = (w == 0) ? Wp : (w == 1) ? Wl : Wr;
    const int lane = f & 31;
    const int c    = (f >> 5) & 3;
    const int t    = f >> 7;
    const int h    = lane >> 4, n = lane & 15;
    v16bf v;
#pragma unroll
    for (int e = 0; e < 16; ++e) {
        const int K = c * 32 + 8 * h + e + (e >= 8 ? 8 : 0);     // runs at 8h and 16+8h
        v[e] = (__bf16)W[(size_t)K * DCH + t * 16 + n];
    }
    pack[(size_t)w * 1024 + f] = v;
}

// A fragment from a f32 row: two contiguous 8-elem runs at kc+8h and kc+16+8h.
__device__ __forceinline__ v16bf load_a_f32(const float* __restrict__ row,
                                            int kc, int h, float scale) {
    const int k0 = kc + 8 * h;
    const float4 fa = *(const float4*)(row + k0);
    const float4 fb = *(const float4*)(row + k0 + 4);
    const float4 fc = *(const float4*)(row + k0 + 16);
    const float4 fd = *(const float4*)(row + k0 + 20);
    v16bf a;
    a[0]  = (__bf16)(fa.x * scale); a[1]  = (__bf16)(fa.y * scale);
    a[2]  = (__bf16)(fa.z * scale); a[3]  = (__bf16)(fa.w * scale);
    a[4]  = (__bf16)(fb.x * scale); a[5]  = (__bf16)(fb.y * scale);
    a[6]  = (__bf16)(fb.z * scale); a[7]  = (__bf16)(fb.w * scale);
    a[8]  = (__bf16)(fc.x * scale); a[9]  = (__bf16)(fc.y * scale);
    a[10] = (__bf16)(fc.z * scale); a[11] = (__bf16)(fc.w * scale);
    a[12] = (__bf16)(fd.x * scale); a[13] = (__bf16)(fd.y * scale);
    a[14] = (__bf16)(fd.z * scale); a[15] = (__bf16)(fd.w * scale);
    return a;
}

// -------------------------------- kernel 1 ---------------------------------
// xp = relu(x @ Wp + bp), output bf16. Grid: 6250 x 256 (8 waves, 1 tile each).
__global__ void k_xp(const float* __restrict__ x, const v16bf* __restrict__ Bp,
                     const float* __restrict__ bp, __bf16* __restrict__ xp) {
    const int lane = threadIdx.x & 31;
    const int wv   = threadIdx.x >> 5;        // column tile 0..7
    const int rb   = blockIdx.x;              // 16-row block
    const int h    = lane >> 4;
    const int ln   = lane & 15;
    const float* xr = x + (size_t)(rb * 16 + ln) * DCH;
    const int cb   = wv * 16;

    v8f acc = {};
#pragma unroll
    for (int c4 = 0; c4 < 4; ++c4) {
        const v16bf a = load_a_f32(xr, c4 * 32, h, 1.0f);
        const v16bf b = Bp[(wv * 4 + c4) * 32 + lane];
        acc = __builtin_amdgcn_wmma_f32_16x16x32_bf16(false, a, false, b,
                                                      (short)0, acc, false, false);
    }
    const float bias = bp[cb + ln];
#pragma unroll
    for (int r = 0; r < 8; ++r) {
        const int m = 8 * h + r;
        const float v = fmaxf(acc[r] + bias, 0.0f);
        xp[(size_t)(rb * 16 + m) * DCH + cb + ln] = (__bf16)v;
    }
}

// -------------------------------- kernel 2 ---------------------------------
// One wave per edge (grid-stride): gather xp[src] row (8B/lane coalesced),
// f32-atomic accumulate into summed[dst]; lane 0 bumps cnt[dst].
__global__ void k_scat(const int* __restrict__ ei, const __bf16* __restrict__ xp,
                       float* __restrict__ summed, float* __restrict__ cnt,
                       int E) {
    const int lane  = threadIdx.x & 31;
    const int wpb   = blockDim.x >> 5;
    const int nwave = gridDim.x * wpb;
    for (int e = blockIdx.x * wpb + (threadIdx.x >> 5); e < E; e += nwave) {
        const int src = ei[e];
        const int dst = ei[E + e];
        const unsigned long long bits =
            *(const unsigned long long*)(xp + (size_t)src * DCH + lane * 4);
        float* sr = summed + (size_t)dst * DCH + lane * 4;
#pragma unroll
        for (int i = 0; i < 4; ++i) {
            const float f = bf16bits_to_f32((unsigned int)(bits >> (16 * i)) & 0xFFFFu);
            __hip_atomic_fetch_add(sr + i, f, __ATOMIC_RELAXED,
                                   __HIP_MEMORY_SCOPE_AGENT);
        }
        if (lane == 0)
            __hip_atomic_fetch_add(cnt + dst, 1.0f, __ATOMIC_RELAXED,
                                   __HIP_MEMORY_SCOPE_AGENT);
    }
}

// -------------------------------- kernel 3 ---------------------------------
__global__ void k_out(const float* __restrict__ summed, const float* __restrict__ cnt,
                      const __bf16* __restrict__ xp,
                      const v16bf* __restrict__ Bl, const float* __restrict__ bl,
                      const v16bf* __restrict__ Br,
                      const float* __restrict__ Wf, const float* __restrict__ bf_,
                      float* __restrict__ out) {
    __shared__ float ssq[16];
    __shared__ float tile[16][DCH];

    const int lane = threadIdx.x & 31;
    const int wv   = threadIdx.x >> 5;
    const int rb   = blockIdx.x;
    const int h    = lane >> 4;
    const int ln   = lane & 15;
    const int arow = rb * 16 + ln;
    const int cb   = wv * 16;

    if (threadIdx.x < 16) ssq[threadIdx.x] = 0.0f;
    __syncthreads();

    const float ic = 1.0f / fmaxf(cnt[arow], 1.0f);   // mean divisor (clip>=1)
    const float*  sr = summed + (size_t)arow * DCH;
    const __bf16* xr = xp + (size_t)arow * DCH;

    v8f acc = {};
#pragma unroll
    for (int c4 = 0; c4 < 4; ++c4) {
        const int kc = c4 * 32;
        const v16bf aA = load_a_f32(sr, kc, h, ic);
        const v8bf  lo = *(const v8bf*)(xr + kc + 8 * h);
        const v8bf  hi = *(const v8bf*)(xr + kc + 16 + 8 * h);
        const v16bf aX = __builtin_shufflevector(lo, hi, 0, 1, 2, 3, 4, 5, 6, 7,
                                                 8, 9, 10, 11, 12, 13, 14, 15);
        const v16bf bL = Bl[(wv * 4 + c4) * 32 + lane];
        const v16bf bR = Br[(wv * 4 + c4) * 32 + lane];
        acc = __builtin_amdgcn_wmma_f32_16x16x32_bf16(false, aA, false, bL,
                                                      (short)0, acc, false, false);
        acc = __builtin_amdgcn_wmma_f32_16x16x32_bf16(false, aX, false, bR,
                                                      (short)0, acc, false, false);
    }

    const float bias = bl[cb + ln];
    float vals[8];
#pragma unroll
    for (int r = 0; r < 8; ++r) {
        const float v = acc[r] + bias;
        vals[r] = v;
        __hip_atomic_fetch_add(&ssq[8 * h + r], v * v, __ATOMIC_RELAXED,
                               __HIP_MEMORY_SCOPE_WORKGROUP);
    }
    __syncthreads();

#pragma unroll
    for (int r = 0; r < 8; ++r) {
        const int m = 8 * h + r;
        const float nrm = fmaxf(sqrtf(ssq[m]), 1e-12f);
        tile[m][cb + ln] = fmaxf(vals[r] / nrm, 0.0f);   // normalize + relu
    }
    __syncthreads();

    // classifier: 96 threads, one (row, class) each; 128-deep dot from LDS
    if (threadIdx.x < 16 * NCLS) {
        const int m   = threadIdx.x / NCLS;
        const int cls = threadIdx.x % NCLS;
        float s = bf_[cls];
#pragma unroll 8
        for (int c = 0; c < DCH; ++c)
            s += tile[m][c] * Wf[c * NCLS + cls];
        out[(size_t)(rb * 16 + m) * NCLS + cls] = s;
    }
}

// ---------------------------------------------------------------------------
extern "C" void kernel_launch(void* const* d_in, const int* in_sizes, int n_in,
                              void* d_out, int out_size, void* d_ws, size_t ws_size,
                              hipStream_t stream) {
    const float* x   = (const float*)d_in[0];
    const int*   ei  = (const int*)d_in[1];     // edge_index (2, E)
    const float* Wp  = (const float*)d_in[2];
    const float* bp  = (const float*)d_in[3];
    const float* Wl  = (const float*)d_in[4];
    const float* bl  = (const float*)d_in[5];
    const float* Wr  = (const float*)d_in[6];
    const float* Wf  = (const float*)d_in[7];
    const float* bf_ = (const float*)d_in[8];
    float* out = (float*)d_out;

    const int E = in_sizes[1] / 2;

    // workspace layout
    char* ws = (char*)d_ws;
    __bf16* xp     = (__bf16*)ws;                               // N*128 bf16
    float*  summed = (float*)(ws + (size_t)NODES * DCH * 2);    // N*128 f32
    float*  cnt    = summed + (size_t)NODES * DCH;              // N f32
    v16bf*  pack   = (v16bf*)(ws + (size_t)NODES * DCH * 2      // 3*32KB bf16
                                 + (size_t)(NODES * DCH + NODES) * sizeof(float));

    // zero the scatter accumulators (summed + cnt contiguous)
    hipMemsetAsync(summed, 0, (size_t)(NODES * DCH + NODES) * sizeof(float), stream);

    k_pack<<<12, 256, 0, stream>>>(Wp, Wl, Wr, pack);

    const int rowBlocks = NODES / 16;   // 6250
    k_xp<<<rowBlocks, 256, 0, stream>>>(x, pack, bp, xp);

    k_scat<<<4096, 256, 0, stream>>>(ei, xp, summed, cnt, E);

    k_out<<<rowBlocks, 256, 0, stream>>>(summed, cnt, xp,
                                         pack + 1024, bl, pack + 2048,
                                         Wf, bf_, out);
}
